// EmotionAwareEncoder_44753559224655
// MI455X (gfx1250) — compile-verified, matrix operand order checked by reading, not run on hardware
//
#include <hip/hip_runtime.h>
#include <hip/hip_bf16.h>
#include <math.h>
#include <stdint.h>

// CDNA5 wave32 WMMA types
typedef __attribute__((ext_vector_type(16))) __bf16 v16bf;
typedef __attribute__((ext_vector_type(8)))  float  v8f;

#define B_     8
#define T_     4096
#define G_     8
#define DG_    64
#define KCODES 1024
#define DMODEL 512

static __device__ __forceinline__ float elu1(float x) {
    return x > 0.f ? x : (__expf(x) - 1.f);
}

// gfx1250 async global->LDS copy (16B), tracked by ASYNCcnt.
// vdst supplies the LDS byte address; vaddr is the 64-bit global address.
static __device__ __forceinline__ void async_g2l_b128(void* lds, const void* gsrc) {
    uint32_t loff = (uint32_t)(uintptr_t)lds;        // flat->LDS: low 32 bits are the LDS offset
    uint64_t ga   = (uint64_t)(uintptr_t)gsrc;
    asm volatile("global_load_async_to_lds_b128 %0, %1, off" :: "v"(loff), "v"(ga) : "memory");
}
static __device__ __forceinline__ void async_wait0() {
    asm volatile("s_wait_asynccnt 0x0" ::: "memory");
}

// ---------------- setup kernels ----------------

__global__ void f2bf_kernel(const float* __restrict__ src, __bf16* __restrict__ dst, int n) {
    int i = blockIdx.x * blockDim.x + threadIdx.x;
    if (i < n) dst[i] = (__bf16)src[i];
}

// head_w (4,512,64) f32 -> (4,64,512) bf16 (so B-matrix K is contiguous per lane)
__global__ void headw_t_kernel(const float* __restrict__ hw, __bf16* __restrict__ hwt) {
    int i = blockIdx.x * blockDim.x + threadIdx.x;
    if (i >= 4 * 512 * 64) return;
    int h = i / (512 * 64);
    int r = i % (512 * 64);
    int d = r / 64;
    int v = r % 64;
    hwt[((size_t)h * 64 + v) * DMODEL + d] = (__bf16)hw[i];
}

// ||codebook||^2 per (g,k)
__global__ void cbnorm_kernel(const float* __restrict__ cb, float* __restrict__ c2) {
    int i = blockIdx.x * blockDim.x + threadIdx.x;
    if (i >= G_ * KCODES) return;
    const float* row = cb + (size_t)i * DG_;
    float s = 0.f;
#pragma unroll
    for (int d = 0; d < DG_; ++d) s += row[d] * row[d];
    c2[i] = s;
}

__global__ void zero_loss_kernel(float* p) {
    if (threadIdx.x == 0 && blockIdx.x == 0) *p = 0.f;
}

// conv1: Cin=1, K=7, +ELU, scalar (tiny FLOPs)
__global__ void conv1_kernel(const float* __restrict__ x, const float* __restrict__ w,
                             const float* __restrict__ bias, __bf16* __restrict__ y) {
    int i = blockIdx.x * blockDim.x + threadIdx.x;
    if (i >= B_ * 64 * T_) return;
    int t = i % T_;
    int c = (i / T_) % 64;
    int b = i / (T_ * 64);
    const float* wr = w + c * 7;
    const float* xr = x + (size_t)b * T_;
    float acc = bias[c];
#pragma unroll
    for (int k = 0; k < 7; ++k) {
        int tt = t - 6 + k;
        if (tt >= 0) acc += wr[k] * xr[tt];
    }
    y[((size_t)b * 64 + c) * T_ + t] = (__bf16)elu1(acc);
}

// ---------------- WMMA causal conv (im2col GEMM) ----------------
// Block: 256 thr = 8 waves. Tile: M=16 Cout x N=128 time (wave -> 16x16).
// K loop: Cin chunks of 32, taps t=0..K-1 each one v_wmma_f32_16x16x32_bf16.
__global__ __launch_bounds__(256) void conv_wmma_kernel(
    const __bf16* __restrict__ X, const __bf16* __restrict__ W,
    const float* __restrict__ bias, __bf16* __restrict__ Y,
    float* __restrict__ Yf, __bf16* __restrict__ Yt,
    int Cin, int Cout, int K, int doElu)
{
    __shared__ __bf16 sX[134][32];      // (128 + maxK-1) time rows x 32 cin (time-major)
    __shared__ __bf16 sW[7][16][32];    // tap x Cout-row x cin

    const int b = blockIdx.z;
    const int mbase = blockIdx.y * 16;
    const int nbase = blockIdx.x * 128;
    const int tid = threadIdx.x;
    const int wave = tid >> 5;
    const int lane = tid & 31;
    const int halo = K - 1;
    const int rows = 128 + halo;
    const int mrow = lane & 15;          // A: M row, B: N column (time)
    const int khalf = (lane >> 4) * 16;  // K half selected by lane group

    v8f acc = {};

    for (int cb = 0; cb < Cin; cb += 32) {
        // stage input tile (causal left-halo, zero pad); channel-major -> time-major transpose
        for (int i = tid; i < rows * 32; i += 256) {
            int lt = i >> 5;
            int c = i & 31;
            int gt = nbase + lt - halo;
            sX[lt][c] = (gt >= 0) ? X[((size_t)b * Cin + cb + c) * T_ + gt] : (__bf16)0.f;
        }
        // stage weight taps for this cin chunk
        for (int i = tid; i < K * 16 * 32; i += 256) {
            int tp = i / 512;
            int r = (i >> 5) & 15;
            int c = i & 31;
            sW[tp][r][c] = W[(((size_t)(mbase + r)) * Cin + cb + c) * K + tp];
        }
        // gfx1250 prefetch of next chunk's activations
        if (cb + 32 < Cin)
            __builtin_prefetch(&X[((size_t)b * Cin + cb + 32 + (tid & 31)) * T_ + nbase], 0, 0);
        __syncthreads();

        for (int t = 0; t < K; ++t) {
            v16bf a  = *(const v16bf*)&sW[t][mrow][khalf];
            v16bf bm = *(const v16bf*)&sX[wave * 16 + mrow + t][khalf];
            acc = __builtin_amdgcn_wmma_f32_16x16x32_bf16(false, a, false, bm,
                                                          (short)0, acc, false, false);
        }
        __syncthreads();
    }

    const int nl = lane & 15;
#pragma unroll
    for (int i = 0; i < 8; ++i) {
        int m = (lane < 16) ? i : (i + 8);   // C/D layout: VGPR i -> M=i (lanes 0-15) / M=i+8
        int c = mbase + m;
        int tt = nbase + wave * 16 + nl;
        float v = acc[i] + bias[c];
        if (doElu) v = elu1(v);
        Y[((size_t)b * Cout + c) * T_ + tt] = (__bf16)v;
        if (Yf) Yf[((size_t)b * Cout + c) * T_ + tt] = v;
        if (Yt) Yt[((size_t)b * T_ + tt) * Cout + c] = (__bf16)v;   // time-major copy
    }
}

// ---------------- VQ: distances via WMMA + argmin ----------------
// Block: (b, g, 128-time tile); each wave owns 16 time rows, scans all 64 code tiles.
// z tile staged with gfx1250 async global->LDS b128 copies (time-major z).
__global__ __launch_bounds__(256) void vq_argmin_kernel(
    const __bf16* __restrict__ Zt, const float* __restrict__ Zf,
    const __bf16* __restrict__ Cb, const float* __restrict__ C2,
    int* __restrict__ Idx, float* __restrict__ aTok)
{
    __shared__ __bf16 sZ[128][64];
    __shared__ float sZ2[128];

    const int b = blockIdx.z;
    const int g = blockIdx.y;
    const int tbase = blockIdx.x * 128;
    const int tid = threadIdx.x;
    const int wave = tid >> 5;
    const int lane = tid & 31;

    // 128 rows x 64 halfs = 1024 x 16B async chunks (8 per row)
    for (int i = tid; i < 1024; i += 256) {
        int r = i >> 3;
        int c = i & 7;
        async_g2l_b128(&sZ[r][c * 8],
                       &Zt[((size_t)b * T_ + tbase + r) * DMODEL + g * DG_ + c * 8]);
    }
    if (tid < 128) {
        float s = 0.f;
        for (int d = 0; d < DG_; ++d) {
            float zv = Zf[((size_t)b * DMODEL + g * DG_ + d) * T_ + tbase + tid];
            s += zv * zv;
        }
        sZ2[tid] = s;
    }
    async_wait0();
    __syncthreads();

    const int nl = lane & 15;
    const int khalf = (lane >> 4) * 16;

    float minv[8];
    int mini[8];
#pragma unroll
    for (int i = 0; i < 8; ++i) { minv[i] = 3.0e38f; mini[i] = 0; }

    for (int tile = 0; tile < KCODES / 16; ++tile) {
        int code = tile * 16 + nl;
        float c2v = C2[g * KCODES + code];
        v8f acc = {};
#pragma unroll
        for (int ks = 0; ks < 2; ++ks) {   // K = 64 = 2 x 32
            v16bf a  = *(const v16bf*)&sZ[wave * 16 + nl][ks * 32 + khalf];
            v16bf bm = *(const v16bf*)&Cb[((size_t)g * KCODES + code) * DG_ + ks * 32 + khalf];
            acc = __builtin_amdgcn_wmma_f32_16x16x32_bf16(false, a, false, bm,
                                                          (short)0, acc, false, false);
        }
#pragma unroll
        for (int i = 0; i < 8; ++i) {
            int m = (lane < 16) ? i : (i + 8);
            float d = sZ2[wave * 16 + m] + c2v - 2.0f * acc[i];
            if (d < minv[i]) { minv[i] = d; mini[i] = code; }
        }
    }
    // reduce over the 16 lanes holding different code columns (wave32, width-16 groups)
#pragma unroll
    for (int i = 0; i < 8; ++i) {
        for (int off = 8; off > 0; off >>= 1) {
            float ov = __shfl_xor(minv[i], off, 16);
            int oi = __shfl_xor(mini[i], off, 16);
            if (ov < minv[i] || (ov == minv[i] && oi < mini[i])) { minv[i] = ov; mini[i] = oi; }
        }
    }
    if (nl == 0) {
#pragma unroll
        for (int i = 0; i < 8; ++i) {
            int m = (lane < 16) ? i : (i + 8);
            int t = tbase + wave * 16 + m;
            Idx[((size_t)b * T_ + t) * G_ + g] = mini[i];
            aTok[(size_t)b * G_ * T_ + (size_t)g * T_ + t] = (float)mini[i];
        }
    }
}

// ---------------- gather q, z_q bf16, vq_loss ----------------
// One wave per (b,t,g); block pre-reduces, one atomicAdd per block.
__global__ __launch_bounds__(256) void vq_apply_kernel(
    const int* __restrict__ Idx, const float* __restrict__ Cbf,
    const float* __restrict__ Zf, __bf16* __restrict__ Zq,
    float* __restrict__ loss)
{
    __shared__ float sAcc;
    if (threadIdx.x == 0) sAcc = 0.f;
    __syncthreads();

    const int tid = threadIdx.x;
    const int wave = tid >> 5;
    const int lane = tid & 31;
    size_t unit = (size_t)blockIdx.x * 8 + wave;   // flat (b*T + t)*G + g
    int g = (int)(unit % G_);
    size_t bt = unit / G_;
    int t = (int)(bt % T_);
    int b = (int)(bt / T_);

    int code = Idx[unit];
    const float* q = Cbf + ((size_t)g * KCODES + code) * DG_;
    float part = 0.f;
#pragma unroll
    for (int s = 0; s < 2; ++s) {
        int d = lane + s * 32;
        float qv = q[d];
        float zv = Zf[((size_t)b * DMODEL + g * DG_ + d) * T_ + t];
        float df = qv - zv;
        part += df * df;
        Zq[((size_t)b * T_ + t) * DMODEL + g * DG_ + d] = (__bf16)qv;
    }
#pragma unroll
    for (int off = 16; off > 0; off >>= 1) part += __shfl_down(part, off, 32);
    if (lane == 0) atomicAdd(&sAcc, part);
    __syncthreads();
    if (tid == 0) atomicAdd(loss, sAcc * (1.0f / ((float)B_ * T_ * DG_)));
}

// ---------------- head projection via WMMA ----------------
// Block: (b, 16-time tile); 8 waves cover (h,vtile) combos, 2 each; K=512.
// z_q tile staged with async global->LDS b128 copies.
__global__ __launch_bounds__(256) void head_wmma_kernel(
    const __bf16* __restrict__ Zq, const __bf16* __restrict__ Hwt,
    const float* __restrict__ Hb, float* __restrict__ Out)
{
    __shared__ __bf16 sA[16][512];
    const int b = blockIdx.y;
    const int tbase = blockIdx.x * 16;
    const int tid = threadIdx.x;
    const int wave = tid >> 5;
    const int lane = tid & 31;

    // 16 rows x 512 halfs = 1024 x 16B async chunks (64 per row)
    for (int i = tid; i < 1024; i += 256) {
        int r = i >> 6;
        int c = i & 63;
        async_g2l_b128(&sA[r][c * 8],
                       &Zq[((size_t)b * T_ + tbase + r) * DMODEL + c * 8]);
    }
    async_wait0();
    __syncthreads();

    const int nl = lane & 15;
    const int khalf = (lane >> 4) * 16;

    for (int s = 0; s < 2; ++s) {
        int combo = wave + s * 8;       // 16 combos: h in [0,4), vtile in [0,4)
        int h = combo >> 2;
        int vt = combo & 3;
        v8f acc = {};
#pragma unroll
        for (int kk = 0; kk < DMODEL; kk += 32) {
            v16bf a  = *(const v16bf*)&sA[nl][kk + khalf];
            v16bf bm = *(const v16bf*)&Hwt[((size_t)h * 64 + vt * 16 + nl) * DMODEL + kk + khalf];
            acc = __builtin_amdgcn_wmma_f32_16x16x32_bf16(false, a, false, bm,
                                                          (short)0, acc, false, false);
        }
#pragma unroll
        for (int i = 0; i < 8; ++i) {
            int m = (lane < 16) ? i : (i + 8);
            int v = vt * 16 + nl;
            float val = acc[i] + Hb[h * 64 + v];
            Out[(((size_t)b * 4 + h) * T_ + (tbase + m)) * 64 + v] = val;
        }
    }
}

// ---------------- host launch ----------------

extern "C" void kernel_launch(void* const* d_in, const int* in_sizes, int n_in,
                              void* d_out, int out_size, void* d_ws, size_t ws_size,
                              hipStream_t stream)
{
    const float* audio = (const float*)d_in[0];
    const float* w1 = (const float*)d_in[1];
    const float* b1 = (const float*)d_in[2];
    const float* w2 = (const float*)d_in[3];
    const float* b2 = (const float*)d_in[4];
    const float* w3 = (const float*)d_in[5];
    const float* b3 = (const float*)d_in[6];
    const float* w4 = (const float*)d_in[7];
    const float* b4 = (const float*)d_in[8];
    const float* w5 = (const float*)d_in[9];
    const float* b5 = (const float*)d_in[10];
    const float* cbf = (const float*)d_in[11];
    const float* hw = (const float*)d_in[12];
    const float* hb = (const float*)d_in[13];

    float* out = (float*)d_out;
    float* aTok = out;                                   // (8,8,4096)
    float* bLog = out + (size_t)B_ * G_ * T_;            // (8,4,4096,64)
    float* loss = bLog + (size_t)B_ * 4 * T_ * 64;       // scalar

    char* ws = (char*)d_ws;
    size_t off = 0;
    auto alloc = [&](size_t bytes) -> void* {
        void* p = ws + off;
        off = (off + bytes + 255) & ~(size_t)255;
        return p;
    };
    __bf16* x1b = (__bf16*)alloc((size_t)B_ * 64 * T_ * 2);
    __bf16* x2b = (__bf16*)alloc((size_t)B_ * 128 * T_ * 2);
    __bf16* x3b = (__bf16*)alloc((size_t)B_ * 256 * T_ * 2);
    __bf16* x4b = (__bf16*)alloc((size_t)B_ * 512 * T_ * 2);
    __bf16* zb  = (__bf16*)alloc((size_t)B_ * 512 * T_ * 2);   // channel-major bf16 z
    __bf16* zbt = (__bf16*)alloc((size_t)B_ * T_ * 512 * 2);   // time-major bf16 z
    float*  zf  = (float*) alloc((size_t)B_ * 512 * T_ * 4);   // channel-major f32 z
    __bf16* zqb = (__bf16*)alloc((size_t)B_ * T_ * 512 * 2);   // time-major bf16 z_q
    int*    idx = (int*)   alloc((size_t)B_ * T_ * G_ * 4);
    __bf16* w2b = (__bf16*)alloc((size_t)128 * 64 * 5 * 2);
    __bf16* w3b = (__bf16*)alloc((size_t)256 * 128 * 5 * 2);
    __bf16* w4b = (__bf16*)alloc((size_t)512 * 256 * 3 * 2);
    __bf16* w5b = (__bf16*)alloc((size_t)512 * 512 * 3 * 2);
    __bf16* cbb = (__bf16*)alloc((size_t)G_ * KCODES * DG_ * 2);
    float*  c2  = (float*) alloc((size_t)G_ * KCODES * 4);
    __bf16* hwt = (__bf16*)alloc((size_t)4 * 64 * 512 * 2);
    (void)ws_size; (void)in_sizes; (void)n_in; (void)out_size;

    auto cvt = [&](const float* s, __bf16* d, int n) {
        f2bf_kernel<<<(n + 255) / 256, 256, 0, stream>>>(s, d, n);
    };
    cvt(w2, w2b, 128 * 64 * 5);
    cvt(w3, w3b, 256 * 128 * 5);
    cvt(w4, w4b, 512 * 256 * 3);
    cvt(w5, w5b, 512 * 512 * 3);
    cvt(cbf, cbb, G_ * KCODES * DG_);
    headw_t_kernel<<<(4 * 512 * 64 + 255) / 256, 256, 0, stream>>>(hw, hwt);
    cbnorm_kernel<<<(G_ * KCODES + 255) / 256, 256, 0, stream>>>(cbf, c2);
    zero_loss_kernel<<<1, 32, 0, stream>>>(loss);

    // layer 1 (scalar)
    conv1_kernel<<<((size_t)B_ * 64 * T_ + 255) / 256, 256, 0, stream>>>(audio, w1, b1, x1b);

    // layers 2..5 (WMMA)
    conv_wmma_kernel<<<dim3(T_ / 128, 128 / 16, B_), 256, 0, stream>>>(
        x1b, w2b, b2, x2b, nullptr, nullptr, 64, 128, 5, 1);
    conv_wmma_kernel<<<dim3(T_ / 128, 256 / 16, B_), 256, 0, stream>>>(
        x2b, w3b, b3, x3b, nullptr, nullptr, 128, 256, 5, 1);
    conv_wmma_kernel<<<dim3(T_ / 128, 512 / 16, B_), 256, 0, stream>>>(
        x3b, w4b, b4, x4b, nullptr, nullptr, 256, 512, 3, 1);
    conv_wmma_kernel<<<dim3(T_ / 128, 512 / 16, B_), 256, 0, stream>>>(
        x4b, w5b, b5, zb, zf, zbt, 512, 512, 3, 0);   // no ELU; keep f32 + time-major z

    // VQ argmin (WMMA distances, async-staged z) -> idx + a_tokens
    vq_argmin_kernel<<<dim3(T_ / 128, G_, B_), 256, 0, stream>>>(zbt, zf, cbb, c2, idx, aTok);

    // gather codebook rows, z_q bf16, vq_loss
    vq_apply_kernel<<<((size_t)B_ * T_ * G_) / 8, 256, 0, stream>>>(idx, cbf, zf, zqb, loss);

    // head logits (WMMA, async-staged z_q)
    head_wmma_kernel<<<dim3(T_ / 16, B_), 256, 0, stream>>>(zqb, hwt, hb, bLog);
}
